// GatedSumLinear_80479097192857
// MI455X (gfx1250) — matrix operand order repaired and checked by reading, not compiled, not run on hardware
//
#include <hip/hip_runtime.h>

// Problem: B=4,S=2048,D=1024,O=1024,H=8  ->  M=8192 tokens, N=1024, K=1024, 8 experts.
// MoE einsum folded into ONE GEMM with K'=H*K=8192:
//   A'[m, h*K+k] = g[m,h]*x[m,k]   (g scale is per-row of A -> per-lane scalar in WMMA A layout)
//   B'[h*K+k, n] = expert_w[h,n,k] (already k-contiguous)
// bf16x3 split (hi/lo) on both operands: ~fp32 accuracy at bf16 WMMA throughput.
// Pipeline: async global->LDS DMA (ASYNCcnt) double-buffers the raw fp32 tiles so the
// WMMA pipes overlap with memory; conversion to bf16 hi/lo happens LDS->LDS.

#define MTOT 8192
#define NTOT 1024
#define KTOT 1024
#define HEXP 8
#define BM   128
#define BN   128
#define BK   32
#define LDSS 40   // padded LDS row stride in bf16 elements (16B-aligned rows, spreads banks)
#define NTILE (HEXP * KTOT / BK)   // 256 K-steps

#define GLOBAL_AS __attribute__((address_space(1)))
#define LDS_AS    __attribute__((address_space(3)))

typedef __attribute__((ext_vector_type(4)))  int    vi4;
typedef __attribute__((ext_vector_type(16))) __bf16 v16bf;
typedef __attribute__((ext_vector_type(8)))  float  v8f;

#if __has_builtin(__builtin_amdgcn_global_load_async_to_lds_b128)
#define HAVE_ASYNC 1
#endif

#if defined(HAVE_ASYNC)
#if __has_builtin(__builtin_amdgcn_s_wait_asynccnt)
#define WAIT_ASYNC(n) __builtin_amdgcn_s_wait_asynccnt(n)
#else
#define WAIT_ASYNC(n) asm volatile("s_wait_asynccnt " #n ::: "memory")
#endif
#else
#define WAIT_ASYNC(n) ((void)0)
#endif

union FragB16 { v16bf v; uint4 q[2]; };

__device__ __forceinline__ unsigned short f2bf(float f) {
  unsigned u = __float_as_uint(f);
  u += 0x7FFFu + ((u >> 16) & 1u);            // round-to-nearest-even
  return (unsigned short)(u >> 16);
}
__device__ __forceinline__ float bf2f(unsigned short b) {
  return __uint_as_float(((unsigned)b) << 16);
}
__device__ __forceinline__ unsigned pack2(unsigned short a, unsigned short b) {
  return (unsigned)a | ((unsigned)b << 16);
}

// ---------------- Kernel 1: gating softmax, writes gT[h][m] (transposed) ----------------
__global__ void gate_softmax_kernel(const float* __restrict__ x,
                                    const float* __restrict__ gw,
                                    const float* __restrict__ gb,
                                    float* __restrict__ gT) {
  const int lane  = threadIdx.x & 31;
  const int wid   = threadIdx.x >> 5;
  const int token = blockIdx.x * 8 + wid;     // one wave32 per token
  const float* xr = x + (size_t)token * KTOT;

  float acc[HEXP];
#pragma unroll
  for (int h = 0; h < HEXP; ++h) acc[h] = 0.f;

  for (int d = lane; d < KTOT; d += 32) {
    const float xv = xr[d];
#pragma unroll
    for (int h = 0; h < HEXP; ++h) acc[h] = fmaf(xv, gw[h * KTOT + d], acc[h]);
  }
#pragma unroll
  for (int h = 0; h < HEXP; ++h) {
#pragma unroll
    for (int off = 16; off >= 1; off >>= 1)
      acc[h] += __shfl_xor(acc[h], off, 32);
  }
  float l[HEXP];
  float mx = acc[0] + gb[0];
  l[0] = mx;
#pragma unroll
  for (int h = 1; h < HEXP; ++h) { l[h] = acc[h] + gb[h]; mx = fmaxf(mx, l[h]); }
  float s = 0.f;
#pragma unroll
  for (int h = 0; h < HEXP; ++h) { l[h] = __expf(l[h] - mx); s += l[h]; }
  const float inv = 1.f / s;
  if (lane == 0) {
#pragma unroll
    for (int h = 0; h < HEXP; ++h) gT[h * MTOT + token] = l[h] * inv;
  }
}

// ---------------- Kernel 2: fused MoE GEMM (K'=8192), async-DMA pipelined bf16x3 WMMA ----
__global__ void __launch_bounds__(256, 1)
moe_wmma_kernel(const float* __restrict__ x,
                const float* __restrict__ ew,
                const float* __restrict__ eb,
                const float* __restrict__ gT,
                float* __restrict__ out) {
  // raw fp32 staging (ping-pong, filled by async DMA) + bf16 hi/lo operand tiles
  __shared__ __align__(16) float rawA[2][BM * BK];
  __shared__ __align__(16) float rawW[2][BN * BK];
  __shared__ __align__(16) unsigned short sAhi[BM * LDSS];
  __shared__ __align__(16) unsigned short sAlo[BM * LDSS];
  __shared__ __align__(16) unsigned short sWhi[BN * LDSS];
  __shared__ __align__(16) unsigned short sWlo[BN * LDSS];

  const int tid   = threadIdx.x;
  const int lane  = tid & 31;
  const int wid   = tid >> 5;
  const int waveM = wid >> 1;                 // 0..3  (4 waves along M)
  const int waveN = wid & 1;                  // 0..1  (2 waves along N)
  const int mBase = blockIdx.y * BM;
  const int nBase = blockIdx.x * BN;

  const v8f vzero = {0.f, 0.f, 0.f, 0.f, 0.f, 0.f, 0.f, 0.f};
  v8f acc[2][4];
#pragma unroll
  for (int mf = 0; mf < 2; ++mf)
#pragma unroll
    for (int nf = 0; nf < 4; ++nf) acc[mf][nf] = vzero;

  // ---- stage issue: 8 x B128 per thread per tile (4 for A, 4 for W) ----
  auto issue_tile = [&](int kk, int buf) {
    const int h = kk >> 10;
    const int k = kk & (KTOT - 1);
#pragma unroll
    for (int i = 0; i < 4; ++i) {
      const int f4  = tid + i * 256;          // 0..1023 float4 slots
      const int row = f4 >> 3;                // 0..127
      const int kc  = (f4 & 7) * 4;           // 0,4,...,28
      const float* gA = x + (size_t)(mBase + row) * KTOT + k + kc;
      const float* gW = ew + (size_t)h * NTOT * KTOT + (size_t)(nBase + row) * KTOT + k + kc;
      float* lA = &rawA[buf][row * BK + kc];
      float* lW = &rawW[buf][row * BK + kc];
#if defined(HAVE_ASYNC)
      __builtin_amdgcn_global_load_async_to_lds_b128(
          (GLOBAL_AS vi4*)gA, (LDS_AS vi4*)lA, 0, 0);
      __builtin_amdgcn_global_load_async_to_lds_b128(
          (GLOBAL_AS vi4*)gW, (LDS_AS vi4*)lW, 0, 0);
#else
      *(float4*)lA = *(const float4*)gA;
      *(float4*)lW = *(const float4*)gW;
#endif
    }
  };

  issue_tile(0, 0);                            // prime the pipeline

  for (int t = 0; t < NTILE; ++t) {
    const int kk  = t * BK;
    const int h   = kk >> 10;
    const int buf = t & 1;

    if (t + 1 < NTILE) {
      issue_tile(kk + BK, buf ^ 1);            // DMA next tile while we work
      WAIT_ASYNC(8);                           // in-order: first 8 (this tile) done
    } else {
      WAIT_ASYNC(0);
    }
    __syncthreads();                           // raw[buf] visible to all waves

    // ---- convert raw fp32 -> bf16 hi/lo tiles (A gets the per-row gate scale) ----
#pragma unroll
    for (int i = 0; i < 4; ++i) {
      const int f4  = tid + i * 256;
      const int row = f4 >> 3;
      const int kc  = (f4 & 7) * 4;

      const float4 xv = *(const float4*)&rawA[buf][row * BK + kc];
      const float  gv = gT[h * MTOT + mBase + row];
      const float a0 = xv.x * gv, a1 = xv.y * gv, a2 = xv.z * gv, a3 = xv.w * gv;
      const unsigned short h0 = f2bf(a0), h1 = f2bf(a1), h2 = f2bf(a2), h3 = f2bf(a3);
      uint2 hp, lp;
      hp.x = pack2(h0, h1);                       hp.y = pack2(h2, h3);
      lp.x = pack2(f2bf(a0 - bf2f(h0)), f2bf(a1 - bf2f(h1)));
      lp.y = pack2(f2bf(a2 - bf2f(h2)), f2bf(a3 - bf2f(h3)));
      *(uint2*)&sAhi[row * LDSS + kc] = hp;
      *(uint2*)&sAlo[row * LDSS + kc] = lp;

      const float4 wv = *(const float4*)&rawW[buf][row * BK + kc];
      const unsigned short w0 = f2bf(wv.x), w1 = f2bf(wv.y), w2 = f2bf(wv.z), w3 = f2bf(wv.w);
      uint2 whp, wlp;
      whp.x = pack2(w0, w1);                      whp.y = pack2(w2, w3);
      wlp.x = pack2(f2bf(wv.x - bf2f(w0)), f2bf(wv.y - bf2f(w1)));
      wlp.y = pack2(f2bf(wv.z - bf2f(w2)), f2bf(wv.w - bf2f(w3)));
      *(uint2*)&sWhi[row * LDSS + kc] = whp;
      *(uint2*)&sWlo[row * LDSS + kc] = wlp;
    }
    __syncthreads();                           // bf16 tiles ready

    // ---- compute: wave tile 32(M) x 64(N), 24 wmma per K-step ----
    // A frag (16-bit, 16x32): lane row = lane%16; K = {k0..k0+7, k0+16..k0+23}, k0 = lane<16?0:8
    const int rA = waveM * 32 + (lane & 15);
    const int k0 = (lane < 16) ? 0 : 8;
    FragB16 ah[2], al[2];
#pragma unroll
    for (int mf = 0; mf < 2; ++mf) {
      const int r = rA + mf * 16;
      ah[mf].q[0] = *(const uint4*)&sAhi[r * LDSS + k0];
      ah[mf].q[1] = *(const uint4*)&sAhi[r * LDSS + k0 + 16];
      al[mf].q[0] = *(const uint4*)&sAlo[r * LDSS + k0];
      al[mf].q[1] = *(const uint4*)&sAlo[r * LDSS + k0 + 16];
    }
    // B frag (16-bit, 32x16): lane col = lane%16; K = {kb0..kb0+7, kb0+16..kb0+23}, kb0 = lane<16?0:8
    const int kb0 = (lane < 16) ? 0 : 8;
#pragma unroll
    for (int nf = 0; nf < 4; ++nf) {
      const int r = waveN * 64 + nf * 16 + (lane & 15);
      FragB16 bh, bl;
      bh.q[0] = *(const uint4*)&sWhi[r * LDSS + kb0];
      bh.q[1] = *(const uint4*)&sWhi[r * LDSS + kb0 + 16];
      bl.q[0] = *(const uint4*)&sWlo[r * LDSS + kb0];
      bl.q[1] = *(const uint4*)&sWlo[r * LDSS + kb0 + 16];
#pragma unroll
      for (int mf = 0; mf < 2; ++mf) {
        v8f c = acc[mf][nf];
        // accumulate low-order terms first: lo*hi + hi*lo + hi*hi
        c = __builtin_amdgcn_wmma_f32_16x16x32_bf16(false, al[mf].v, false, bh.v, (short)0, c, false, false);
        c = __builtin_amdgcn_wmma_f32_16x16x32_bf16(false, ah[mf].v, false, bl.v, (short)0, c, false, false);
        c = __builtin_amdgcn_wmma_f32_16x16x32_bf16(false, ah[mf].v, false, bh.v, (short)0, c, false, false);
        acc[mf][nf] = c;
      }
    }
    __syncthreads();                           // done reading bf16 tiles before next convert
  }

  // ---- epilogue: add expert-bias term sum_h g[m,h]*eb[h,n], write fp32 out ----
#pragma unroll
  for (int nf = 0; nf < 4; ++nf) {
    const int n = nBase + waveN * 64 + nf * 16 + (lane & 15);
    float ebv[HEXP];
#pragma unroll
    for (int hh = 0; hh < HEXP; ++hh) ebv[hh] = eb[hh * NTOT + n];
#pragma unroll
    for (int mf = 0; mf < 2; ++mf) {
#pragma unroll
      for (int j = 0; j < 8; ++j) {
        // C/D layout: VGPR j holds M = j (lanes 0-15) or j+8 (lanes 16-31), N = lane%16
        const int m = mBase + waveM * 32 + mf * 16 + j + ((lane >= 16) ? 8 : 0);
        float b = 0.f;
#pragma unroll
        for (int hh = 0; hh < HEXP; ++hh) b = fmaf(gT[hh * MTOT + m], ebv[hh], b);
        out[(size_t)m * NTOT + n] = acc[mf][nf][j] + b;
      }
    }
  }
}

extern "C" void kernel_launch(void* const* d_in, const int* in_sizes, int n_in,
                              void* d_out, int out_size, void* d_ws, size_t ws_size,
                              hipStream_t stream) {
  const float* x  = (const float*)d_in[0];   // [4,2048,1024]
  const float* gw = (const float*)d_in[1];   // [8,1024]
  const float* gb = (const float*)d_in[2];   // [8]
  const float* ew = (const float*)d_in[3];   // [8,1024,1024]
  const float* eb = (const float*)d_in[4];   // [8,1024]
  float* out = (float*)d_out;                // [8192,1024]
  float* gT  = (float*)d_ws;                 // [8,8192] transposed gate probs (256 KB)

  gate_softmax_kernel<<<dim3(MTOT / 8), dim3(256), 0, stream>>>(x, gw, gb, gT);
  moe_wmma_kernel<<<dim3(NTOT / BN, MTOT / BM), dim3(256), 0, stream>>>(x, ew, eb, gT, out);
}
